// FCLayer_7885559955602
// MI455X (gfx1250) — compile-verified, matrix operand order checked by reading, not run on hardware
//
#include <hip/hip_runtime.h>

// ---------------------------------------------------------------------------
// Fused 4-layer MLP for MI455X (gfx1250), f16 WMMA path with async-to-LDS
// weight staging (double-buffered, ASYNCcnt-tracked).
//   L0: block-diag 50 x Linear(16,16)      (no activation)
//   L1: 800->800 + tanh                    (weights staged via async copy)
//   L2: 800->800 + tanh                    (weights staged via async copy)
//   L3: 800->2
// Weights converted once to f16 in d_ws; activations live in LDS only.
// ---------------------------------------------------------------------------

typedef __attribute__((ext_vector_type(16))) _Float16 v16h;
typedef __attribute__((ext_vector_type(8)))  _Float16 h8;
typedef __attribute__((ext_vector_type(4)))  _Float16 h4;
typedef __attribute__((ext_vector_type(8)))  float    v8f;

#define ND         800    // feature dim
#define NSLOT      50     // 16-wide slots
#define MTILE      64     // batch rows per workgroup
#define NTHREADS   128    // 4 waves (wave32)
#define LDS_STRIDE 808    // halves per LDS row (800 + 8 pad)

// LDS layout (bytes from dynamic-LDS base)
#define ACT_BYTES   (MTILE * LDS_STRIDE * 2)        // 103,424 per buffer
#define WSTAGE_HALVES (32 * LDS_STRIDE)             // one 32-row tile pair
#define WSTAGE_BYTES  (WSTAGE_HALVES * 2)           // 51,712 per buffer
#define LDS_TOTAL   (2 * ACT_BYTES + 2 * WSTAGE_BYTES)  // 310,272 <= 320K

// ws layout (in halves)
#define WS_W0   0                         // [50][16][16]
#define WS_W1   (WS_W0 + NSLOT*16*16)     // [800][800]
#define WS_W2   (WS_W1 + ND*ND)           // [800][800]
#define WS_W3   (WS_W2 + ND*ND)           // [16][800]  (rows 2..15 zero)
#define WS_TOT  (WS_W3 + 16*ND)           // 1,305,600 halves

// ---------------------------------------------------------------------------
// Prep: fp32 weights -> f16 in workspace (W3 zero-padded to 16 rows)
// ---------------------------------------------------------------------------
__global__ void __launch_bounds__(256)
mlp_prep_f16(const float* __restrict__ W0, const float* __restrict__ W1,
             const float* __restrict__ W2, const float* __restrict__ W3,
             _Float16* __restrict__ ws) {
  const int n0 = NSLOT * 16 * 16;          // 12800
  const int n1 = ND * ND;                  // 640000
  for (long i = (long)blockIdx.x * blockDim.x + threadIdx.x; i < (long)WS_TOT;
       i += (long)gridDim.x * blockDim.x) {
    long j = i;
    if (j < n0) { ws[WS_W0 + j] = (_Float16)W0[j]; continue; }
    j -= n0;
    if (j < n1) { ws[WS_W1 + j] = (_Float16)W1[j]; continue; }
    j -= n1;
    if (j < n1) { ws[WS_W2 + j] = (_Float16)W2[j]; continue; }
    j -= n1;
    long row = j / ND, col = j % ND;
    ws[WS_W3 + j] = (row < 2) ? (_Float16)W3[row * ND + col] : (_Float16)0.0f;
  }
}

// ---------------------------------------------------------------------------
// CDNA5 async copy: 16B global -> LDS per lane, tracked with ASYNCcnt.
// ---------------------------------------------------------------------------
__device__ __forceinline__ void async_b128_to_lds(unsigned lds_byte_off,
                                                  const void* gaddr) {
  asm volatile("global_load_async_to_lds_b128 %0, %1, off"
               :: "v"(lds_byte_off), "v"(gaddr)
               : "memory");
}
__device__ __forceinline__ void wait_async_le25() {
  asm volatile("s_wait_asynccnt 0x19" ::: "memory");
}
__device__ __forceinline__ void wait_async_0() {
  asm volatile("s_wait_asynccnt 0x0" ::: "memory");
}

// Stage a 32-row x 800-col f16 weight tile-pair into LDS [32][LDS_STRIDE].
// 3200 16B chunks / 128 threads = 25 async issues per thread (ASYNCcnt <= 50).
__device__ __forceinline__ void stage_pair_async(const _Float16* __restrict__ Wg,
                                                 unsigned wb_byte_off, int tid) {
#pragma unroll
  for (int i = 0; i < 25; ++i) {
    const int c   = i * NTHREADS + tid;
    const int row = c / 100;          // 100 chunks per 800-half row
    const int col = c % 100;
    async_b128_to_lds(wb_byte_off + (unsigned)(row * LDS_STRIDE + col * 8) * 2,
                      (const void*)(Wg + row * ND + col * 8));
  }
}

// ---------------------------------------------------------------------------
// Fragment loads (ISA 7.12.2 16-bit layouts), both act and weight-stage
// buffers use LDS_STRIDE rows.
// A (16x32, MxK): lane m=L&15, hk=L>>4. halves[0..7] <- K = hk*8 + 0..7,
//                 halves[8..15] <- K = 16 + hk*8 + 0..7.
// B (32x16, KxN) of y = x*W^T with row-major W[N][K]: lane n=L&15, same
//                 per-lane K pattern -> two contiguous 16B loads of W row n.
// ---------------------------------------------------------------------------
__device__ __forceinline__ v16h frag_ld(const _Float16* buf, int rowBase,
                                        int lane, int k0) {
  const int r  = rowBase + (lane & 15);
  const int hk = (lane >> 4) & 1;
  const _Float16* p = buf + r * LDS_STRIDE + k0 + hk * 8;
  v16h f;
  ((h8*)&f)[0] = *(const h8*)(p);
  ((h8*)&f)[1] = *(const h8*)(p + 16);
  return f;
}

// fragment with K=16..31 forced to zero (block-diagonal layer 0)
__device__ __forceinline__ v16h frag_ld_k16(const _Float16* buf, int rowBase,
                                            int lane, int k0) {
  const int r  = rowBase + (lane & 15);
  const int hk = (lane >> 4) & 1;
  v16h f;
  ((h8*)&f)[0] = *(const h8*)(buf + r * LDS_STRIDE + k0 + hk * 8);
  h8 z = {};
  ((h8*)&f)[1] = z;
  return f;
}

// B fragment for K=16 weights (row-major ld=16, global), K=16..31 zero
__device__ __forceinline__ v16h frag_b_w_k16(const _Float16* __restrict__ W,
                                             int lane) {
  const int n  = lane & 15;
  const int hk = (lane >> 4) & 1;
  v16h b;
  ((h8*)&b)[0] = *(const h8*)(W + n * 16 + hk * 8);
  h8 z = {};
  ((h8*)&b)[1] = z;
  return b;
}

// B fragment from global row-major weights (ldw halves)
__device__ __forceinline__ v16h frag_b_w(const _Float16* __restrict__ W,
                                         int nBase, int lane, int k0, int ldw) {
  const int n  = nBase + (lane & 15);
  const int hk = (lane >> 4) & 1;
  const _Float16* p = W + (long)n * ldw + k0 + hk * 8;
  v16h b;
  ((h8*)&b)[0] = *(const h8*)(p);
  ((h8*)&b)[1] = *(const h8*)(p + 16);
  return b;
}

// D tile (16x16 f32): VGPR r, lane L -> row = r + 8*(L>>4), col = L&15
__device__ __forceinline__ void store_tile_lds(_Float16* buf, int mBase,
                                               int nBase, int lane, v8f acc,
                                               float bias, bool do_tanh) {
  const int n  = nBase + (lane & 15);
  const int hk = (lane >> 4) & 1;
#pragma unroll
  for (int r = 0; r < 8; ++r) {
    float v = acc[r] + bias;
    if (do_tanh) v = tanhf(v);
    buf[(mBase + r + 8 * hk) * LDS_STRIDE + n] = (_Float16)v;
  }
}

__device__ __forceinline__ v8f wmma_f16(v16h a, v16h b, v8f c) {
  return __builtin_amdgcn_wmma_f32_16x16x32_f16(
      /*neg_a=*/false, a, /*neg_b=*/false, b,
      /*c_mod=*/(short)0, c, /*reuse_a=*/false, /*reuse_b=*/false);
}

// ---------------------------------------------------------------------------
// Dense 800->800 layer, async-staged weights (double buffer), A-frag reuse
// across the two N-tiles of each staged pair.
// ---------------------------------------------------------------------------
__device__ __forceinline__ void dense_layer_staged(
    const _Float16* src, _Float16* dst, _Float16* wstage,
    const _Float16* __restrict__ Wh, const float* __restrict__ bias,
    int mBase, int lane, int tid) {
  const unsigned wbOff0 = (unsigned)(2 * ACT_BYTES);
  const unsigned wbOff1 = wbOff0 + WSTAGE_BYTES;

  // prologue: stage pair 0
  stage_pair_async(Wh, wbOff0, tid);

  for (int p = 0; p < NSLOT / 2; ++p) {
    if (p + 1 < NSLOT / 2) {
      if (p > 0) __syncthreads();   // pair p-1 consumers done -> buffer free
      stage_pair_async(Wh + (size_t)(p + 1) * 32 * ND,
                       ((p + 1) & 1) ? wbOff1 : wbOff0, tid);
      wait_async_le25();            // batch p landed (in-order completion)
    } else {
      wait_async_0();               // last pair: drain
    }
    __syncthreads();                // all waves' share of pair p visible

    const _Float16* wb = wstage + (size_t)(p & 1) * WSTAGE_HALVES;
    v8f acc0 = {}, acc1 = {};
#pragma unroll 5
    for (int kt = 0; kt < ND / 32; ++kt) {
      v16h a  = frag_ld(src, mBase, lane, kt * 32);
      v16h b0 = frag_ld(wb, 0, lane, kt * 32);
      v16h b1 = frag_ld(wb, 16, lane, kt * 32);
      acc0 = wmma_f16(a, b0, acc0);
      acc1 = wmma_f16(a, b1, acc1);
    }
    store_tile_lds(dst, mBase, (2 * p) * 16, lane, acc0,
                   bias[(2 * p) * 16 + (lane & 15)], /*tanh=*/true);
    store_tile_lds(dst, mBase, (2 * p + 1) * 16, lane, acc1,
                   bias[(2 * p + 1) * 16 + (lane & 15)], /*tanh=*/true);
  }
}

// ---------------------------------------------------------------------------
// Fused MLP kernel: 1 workgroup = 64 batch rows, 4 waves, 1 wave = 16 rows.
// ---------------------------------------------------------------------------
__global__ void __launch_bounds__(NTHREADS)
mlp_fused(const float* __restrict__ x, const _Float16* __restrict__ ws,
          const float* __restrict__ b0, const float* __restrict__ b1,
          const float* __restrict__ b2, const float* __restrict__ b3,
          float* __restrict__ out) {
  extern __shared__ _Float16 smem[];
  _Float16* bufA   = smem;                           // [MTILE][LDS_STRIDE]
  _Float16* bufB   = smem + MTILE * LDS_STRIDE;      // [MTILE][LDS_STRIDE]
  _Float16* wstage = smem + 2 * MTILE * LDS_STRIDE;  // 2 x [32][LDS_STRIDE]

  const int tid   = threadIdx.x;
  const int lane  = tid & 31;
  const int wave  = tid >> 5;
  const int mBase = wave * 16;
  const long rowBase = (long)blockIdx.x * MTILE;

  const _Float16* W0h = ws + WS_W0;
  const _Float16* W1h = ws + WS_W1;
  const _Float16* W2h = ws + WS_W2;
  const _Float16* W3h = ws + WS_W3;

  // ---- stage x tile (f32 -> f16) into bufA, cooperatively ----
  {
    const float4* xv = (const float4*)(x + rowBase * ND);
    const int nchunk = MTILE * (ND / 4);  // 12800 float4s
    for (int c = tid; c < nchunk; c += NTHREADS) {
      const int r  = c / (ND / 4);
      const int c4 = c % (ND / 4);
      float4 f = xv[c];
      h4 hv = {(_Float16)f.x, (_Float16)f.y, (_Float16)f.z, (_Float16)f.w};
      *(h4*)(bufA + r * LDS_STRIDE + c4 * 4) = hv;
    }
  }
  __syncthreads();

  // ---- L0: block-diagonal 50 x (16x16), K padded to 32 with zeros ----
  for (int s = 0; s < NSLOT; ++s) {
    v16h a = frag_ld_k16(bufA, mBase, lane, s * 16);
    v16h b = frag_b_w_k16(W0h + s * 256, lane);
    v8f c = {};
    c = wmma_f16(a, b, c);
    store_tile_lds(bufB, mBase, s * 16, lane, c, b0[s * 16 + (lane & 15)],
                   /*tanh=*/false);
  }
  __syncthreads();

  // ---- L1: bufB -> bufA, tanh (async-staged weights) ----
  dense_layer_staged(bufB, bufA, wstage, W1h, b1, mBase, lane, tid);
  __syncthreads();

  // ---- L2: bufA -> bufB, tanh (async-staged weights) ----
  dense_layer_staged(bufA, bufB, wstage, W2h, b2, mBase, lane, tid);
  __syncthreads();

  // ---- L3: 800 -> 2 via padded [16][800] weight, write f32 out ----
  {
    v8f acc = {};
#pragma unroll 5
    for (int kt = 0; kt < ND / 32; ++kt) {
      v16h a = frag_ld(bufB, mBase, lane, kt * 32);
      v16h b = frag_b_w(W3h, 0, lane, kt * 32, ND);
      acc = wmma_f16(a, b, acc);
    }
    const int n  = lane & 15;
    const int hk = (lane >> 4) & 1;
    if (n < 2) {
      const float bias = b3[n];
#pragma unroll
      for (int r = 0; r < 8; ++r) {
        out[(rowBase + mBase + r + 8 * hk) * 2 + n] = acc[r] + bias;
      }
    }
  }
}

// ---------------------------------------------------------------------------
extern "C" void kernel_launch(void* const* d_in, const int* in_sizes, int n_in,
                              void* d_out, int out_size, void* d_ws,
                              size_t ws_size, hipStream_t stream) {
  const float* x  = (const float*)d_in[0];
  const float* W0 = (const float*)d_in[1];
  const float* b0 = (const float*)d_in[2];
  const float* W1 = (const float*)d_in[3];
  const float* b1 = (const float*)d_in[4];
  const float* W2 = (const float*)d_in[5];
  const float* b2 = (const float*)d_in[6];
  const float* W3 = (const float*)d_in[7];
  const float* b3 = (const float*)d_in[8];
  _Float16* ws = (_Float16*)d_ws;

  // 1) weights -> f16 workspace
  mlp_prep_f16<<<640, 256, 0, stream>>>(W0, W1, W2, W3, ws);

  // 2) fused MLP
  const int batch  = in_sizes[0] / ND;          // 65536
  const int nblock = batch / MTILE;             // 1024
  const size_t shmem = (size_t)LDS_TOTAL;       // 310,272 B (<= 320K WGP LDS)
  hipFuncSetAttribute((const void*)mlp_fused,
                      hipFuncAttributeMaxDynamicSharedMemorySize, (int)shmem);
  mlp_fused<<<nblock, NTHREADS, shmem, stream>>>(
      x, ws, b0, b1, b2, b3, (float*)d_out);
}
